// SelfAttention_22222160789906
// MI455X (gfx1250) — matrix operand order repaired and checked by reading, not compiled, not run on hardware
//
#include <hip/hip_runtime.h>
#include <hip/hip_bf16.h>

// ---------------------------------------------------------------------------
// Self-attention (B=4, S=2048, D=1024), causal, softmax scale 1/sqrt(1024).
// Compute-bound (~120 GFLOP vs ~80MB traffic) -> BF16 WMMA path:
//   stage 0: fp32 -> bf16 convert of X, Wq, Wk, Wv into workspace
//   stage 1: WMMA GEMM  Q = X Wq^T, K = X Wk^T, Vt = (X Wv^T)^T  (bf16 out)
//            LDS staging via CDNA5 global_load_async_to_lds_b128 (ASYNCcnt),
//            double-buffered so the async DMA overlaps the WMMA pipe.
//   stage 2: flash attention with v_wmma_f32_16x16x32_bf16 for QK^T and PV
// ---------------------------------------------------------------------------

typedef __attribute__((ext_vector_type(16))) __bf16        v16bf;
typedef __attribute__((ext_vector_type(8)))  float         v8f;
typedef __attribute__((ext_vector_type(4)))  unsigned int  u32x4;
typedef __attribute__((ext_vector_type(8)))  unsigned int  u32x8;

static constexpr int BATCH = 4;
static constexpr int SEQ   = 2048;
static constexpr int DIM   = 1024;

__device__ __forceinline__ unsigned short f2bf(float f) {
  union { float f; unsigned int u; } c; c.f = f;
  unsigned int u = c.u;
  // round-to-nearest-even
  unsigned int r = (u + 0x7FFFu + ((u >> 16) & 1u)) >> 16;
  return (unsigned short)r;
}

// Build a 16xbf16 fragment (32 bytes) from two 16-byte chunks.
__device__ __forceinline__ v16bf load_frag2(const unsigned short* p0,
                                            const unsigned short* p1) {
  u32x4 lo = *(const u32x4*)p0;
  u32x4 hi = *(const u32x4*)p1;
  u32x8 w  = __builtin_shufflevector(lo, hi, 0, 1, 2, 3, 4, 5, 6, 7);
  return __builtin_bit_cast(v16bf, w);
}

// CDNA5 async copy: 16B global -> LDS per lane, tracked with ASYNCcnt.
// LDS dest VGPR takes the wave-relative LDS byte offset == low 32 bits of a
// generic pointer to a __shared__ object.
__device__ __forceinline__ void async_g2l_b128(void* lds_ptr,
                                               const void* gptr) {
  unsigned int lds_off = (unsigned int)(unsigned long long)lds_ptr;
  unsigned long long ga = (unsigned long long)gptr;
  asm volatile("global_load_async_to_lds_b128 %0, %1, off"
               :: "v"(lds_off), "v"(ga) : "memory");
}
__device__ __forceinline__ void wait_async0() {
  asm volatile("s_wait_asynccnt 0" ::: "memory");
}

// ---------------------------------------------------------------------------
// Stage 0: fp32 -> bf16
// ---------------------------------------------------------------------------
__global__ void convert_bf16(const float* __restrict__ src,
                             unsigned short* __restrict__ dst, size_t n) {
  size_t i = (size_t)blockIdx.x * blockDim.x + threadIdx.x;
  size_t stride = (size_t)gridDim.x * blockDim.x;
  for (; i < n; i += stride) dst[i] = f2bf(src[i]);
}

// ---------------------------------------------------------------------------
// Stage 1: QKV projection GEMM.
//   Y[n,o] = sum_d Xbf[n,d] * W[o,d]   (n = b*S+s, 8192 rows)
//   z = blockIdx.z selects Wq/Wk/Wv.  z<2 -> row-major out; z==2 -> Vt[b,o,s].
//   Block tile 128(M) x 128(N), BK=32; 8 waves as 4x2, each wave 32x64
//   (2 A-frags x 4 B-frags -> 8 WMMA per K-step).  Double-buffered async
//   global->LDS staging: fill of step k+1 overlaps compute of step k.
// ---------------------------------------------------------------------------
__global__ __launch_bounds__(256) void qkv_gemm(
    const unsigned short* __restrict__ Xbf,
    const unsigned short* __restrict__ Wbf,
    unsigned short* __restrict__ Qp,
    unsigned short* __restrict__ Kp,
    unsigned short* __restrict__ Vtp) {
  constexpr int BM = 128, BN = 128, BK = 32;
  constexpr int NK = DIM / BK;
  constexpr int LDT = 40;  // ushort row stride (80B, multiple of 16B)
  __shared__ unsigned short ldsA[2][BM * LDT];
  __shared__ unsigned short ldsB[2][BN * LDT];

  const int z = blockIdx.z;
  const unsigned short* W = Wbf + (size_t)z * DIM * DIM;
  const int rowBase = blockIdx.x * BM;
  const int colBase = blockIdx.y * BN;
  const int tid  = threadIdx.x;
  const int wave = tid >> 5;
  const int lane = tid & 31;
  const int lhi  = lane >> 4;   // 0/1 half-wave (wave32 WMMA layout)
  const int llo  = lane & 15;
  const int wm   = wave & 3;    // 4 waves along M (32 rows each)
  const int wn   = wave >> 2;   // 2 waves along N (64 cols each)

  v8f acc[2][4] = {};

  const int fr = tid >> 2;      // fill: thread -> (row, 16B chunk)
  const int fc = tid & 3;

  // 4 async b128 issues per thread per fill (A + B tiles, 2 row groups)
  auto fill = [&](int bufi, int k0) {
#pragma unroll
    for (int i = 0; i < 2; ++i) {
      int row = fr + i * 64;
      async_g2l_b128(&ldsA[bufi][row * LDT + fc * 8],
                     Xbf + (size_t)(rowBase + row) * DIM + k0 + fc * 8);
      async_g2l_b128(&ldsB[bufi][row * LDT + fc * 8],
                     W + (size_t)(colBase + row) * DIM + k0 + fc * 8);
    }
  };

  // prologue: stage first tile
  fill(0, 0);
  wait_async0();
  __syncthreads();

  for (int ki = 0; ki < NK; ++ki) {
    const int cur = ki & 1;
    const bool more = (ki + 1 < NK);
    if (more) fill(cur ^ 1, (ki + 1) * BK);  // overlaps compute below

    // A fragments: lane holds row M=llo; halfs 0..7 at K=lhi*8.., 8..15 at +16
    v16bf afrag[2];
#pragma unroll
    for (int tm = 0; tm < 2; ++tm) {
      const unsigned short* p =
          &ldsA[cur][(wm * 32 + tm * 16 + llo) * LDT + lhi * 8];
      afrag[tm] = load_frag2(p, p + 16);
    }
    // B fragments: lane holds col N=llo; halfs 0..7 at K=lhi*8.., 8..15 at +16
    v16bf bfrag[4];
#pragma unroll
    for (int tn = 0; tn < 4; ++tn) {
      const unsigned short* p =
          &ldsB[cur][(wn * 64 + tn * 16 + llo) * LDT + lhi * 8];
      bfrag[tn] = load_frag2(p, p + 16);
    }
#pragma unroll
    for (int tm = 0; tm < 2; ++tm)
#pragma unroll
      for (int tn = 0; tn < 4; ++tn)
        acc[tm][tn] = __builtin_amdgcn_wmma_f32_16x16x32_bf16(
            false, afrag[tm], false, bfrag[tn], (short)0, acc[tm][tn],
            false, false);

    if (more) wait_async0();  // drain remaining DMA latency after compute
    __syncthreads();
  }

  // Branch-free epilogue: per-tile base + row stride.
  // z<2: dst[row*DIM + col], stride DIM.  z==2: Vt[b*D*S + col*S + s],
  // consecutive rows are consecutive s -> stride 1.
  unsigned short* dst = (z == 0) ? Qp : ((z == 1) ? Kp : Vtp);
  const bool transposed = (z == 2);
#pragma unroll
  for (int tm = 0; tm < 2; ++tm)
#pragma unroll
    for (int tn = 0; tn < 4; ++tn) {
      int col  = colBase + wn * 64 + tn * 16 + llo;
      int rowb = rowBase + wm * 32 + tm * 16 + 8 * lhi;
      size_t base, rstride;
      if (!transposed) {
        base = (size_t)rowb * DIM + col;
        rstride = DIM;
      } else {
        int bb = rowb >> 11;           // /SEQ (block rows never straddle batch)
        int ss = rowb & (SEQ - 1);
        base = (size_t)bb * DIM * SEQ + (size_t)col * SEQ + ss;
        rstride = 1;
      }
#pragma unroll
      for (int r = 0; r < 8; ++r)
        dst[base + (size_t)r * rstride] = f2bf(acc[tm][tn][r]);
    }
}

// ---------------------------------------------------------------------------
// Stage 2: causal flash attention. One block = 16 query rows of one batch.
//  - 8 waves split the 1024-dim QK^T reduction (128 dims each); partial
//    16x32 score tiles reduced across waves via ds_add_f32 atomics in LDS.
//  - wave-group softmax bookkeeping (running max m, sum l, rescale alpha).
//  - each wave owns 128 output dims: PV via WMMA, B-frags contiguous from Vt.
// ---------------------------------------------------------------------------
__global__ __launch_bounds__(256) void flash_attn(
    const unsigned short* __restrict__ Q,
    const unsigned short* __restrict__ K,
    const unsigned short* __restrict__ Vt,
    float* __restrict__ out) {
  constexpr float SCALE   = 0.03125f;  // 1/sqrt(1024)
  constexpr float NEG_BIG = -3.0e38f;

  const int b     = blockIdx.x / (SEQ / 16);
  const int qBase = (blockIdx.x % (SEQ / 16)) * 16;
  const int tid  = threadIdx.x;
  const int wave = tid >> 5;
  const int lane = tid & 31;
  const int lhi  = lane >> 4;
  const int llo  = lane & 15;

  __shared__ float          scLDS[16][36];  // fp32 score tile (padded)
  __shared__ unsigned short pLDS[16][40];   // bf16 probabilities (padded)
  __shared__ float          mLDS[16], lLDS[16], aLDS[16];

  if (tid < 16) { mLDS[tid] = NEG_BIG; lLDS[tid] = 0.0f; }

  // Preload Q fragments for this wave's 128-dim slice (4 x 16x32 A-frags).
  v16bf qf[4];
  {
    const unsigned short* qrow = Q + (size_t)(b * SEQ + qBase + llo) * DIM;
#pragma unroll
    for (int kk = 0; kk < 4; ++kk) {
      const unsigned short* p = qrow + wave * 128 + kk * 32 + lhi * 8;
      qf[kk] = load_frag2(p, p + 16);
    }
  }

  v8f acc[8] = {};  // 16 rows x 128 dims of O per wave

  const int nkb = qBase / 32 + 1;  // causal: key blocks of 32
  for (int kb = 0; kb < nkb; ++kb) {
    const int keyBase = kb * 32;

    // prefetch next key block's K rows for this wave's dim slice
    if (kb + 1 < nkb) {
      const unsigned short* nk =
          K + (size_t)(b * SEQ + keyBase + 32 + lane) * DIM + wave * 128;
      __builtin_prefetch(nk, 0, 0);
    }

    // zero score tile
    for (int i = tid; i < 16 * 36; i += 256)
      (&scLDS[0][0])[i] = 0.0f;
    __syncthreads();

    // partial QK^T over this wave's 128 dims -> two 16x16 tiles
#pragma unroll
    for (int t = 0; t < 2; ++t) {
      v8f sc = {};
      const unsigned short* krow =
          K + (size_t)(b * SEQ + keyBase + t * 16 + llo) * DIM;
#pragma unroll
      for (int kk = 0; kk < 4; ++kk) {
        const unsigned short* p = krow + wave * 128 + kk * 32 + lhi * 8;
        v16bf kf = load_frag2(p, p + 16);
        sc = __builtin_amdgcn_wmma_f32_16x16x32_bf16(
            false, qf[kk], false, kf, (short)0, sc, false, false);
      }
#pragma unroll
      for (int r = 0; r < 8; ++r)
        atomicAdd(&scLDS[r + 8 * lhi][t * 16 + llo], sc[r]);
    }
    __syncthreads();

    // online softmax (one thread per query row)
    if (tid < 16) {
      const int r  = tid;
      const int qi = qBase + r;
      int nv = qi - keyBase + 1;           // causal valid count (>=1)
      if (nv > 32) nv = 32;
      float m_old = mLDS[r];
      float vmax  = m_old;
      for (int c = 0; c < nv; ++c)
        vmax = fmaxf(vmax, scLDS[r][c] * SCALE);
      float alpha = __expf(m_old - vmax);
      float sum = 0.0f;
      for (int c = 0; c < 32; ++c) {
        float p = (c < nv) ? __expf(scLDS[r][c] * SCALE - vmax) : 0.0f;
        sum += p;
        pLDS[r][c] = f2bf(p);
      }
      mLDS[r] = vmax;
      lLDS[r] = lLDS[r] * alpha + sum;
      aLDS[r] = alpha;
    }
    __syncthreads();

    // rescale accumulators and add P x V for this wave's 128 dims
    float a8[8];
#pragma unroll
    for (int r = 0; r < 8; ++r) a8[r] = aLDS[r + 8 * lhi];
#pragma unroll
    for (int nt = 0; nt < 8; ++nt)
#pragma unroll
      for (int r = 0; r < 8; ++r) acc[nt][r] *= a8[r];

    const unsigned short* pp = &pLDS[llo][lhi * 8];
    v16bf pf = load_frag2(pp, pp + 16);
#pragma unroll
    for (int nt = 0; nt < 8; ++nt) {
      int n = wave * 128 + nt * 16 + llo;
      const unsigned short* vrow =
          Vt + (size_t)b * DIM * SEQ + (size_t)n * SEQ + keyBase + lhi * 8;
      v16bf vf = load_frag2(vrow, vrow + 16);
      acc[nt] = __builtin_amdgcn_wmma_f32_16x16x32_bf16(
          false, pf, false, vf, (short)0, acc[nt], false, false);
    }
    __syncthreads();
  }

  // epilogue: divide by running sum, store fp32
#pragma unroll
  for (int nt = 0; nt < 8; ++nt) {
    int col = wave * 128 + nt * 16 + llo;
#pragma unroll
    for (int r = 0; r < 8; ++r) {
      int row = qBase + r + 8 * lhi;
      out[((size_t)b * SEQ + row) * DIM + col] = acc[nt][r] / lLDS[r + 8 * lhi];
    }
  }
}

// ---------------------------------------------------------------------------
// Host launcher
// ---------------------------------------------------------------------------
extern "C" void kernel_launch(void* const* d_in, const int* in_sizes, int n_in,
                              void* d_out, int out_size, void* d_ws,
                              size_t ws_size, hipStream_t stream) {
  (void)in_sizes; (void)n_in; (void)out_size; (void)ws_size;
  const float* X  = (const float*)d_in[0];
  const float* wq = (const float*)d_in[1];
  const float* wk = (const float*)d_in[2];
  const float* wv = (const float*)d_in[3];

  const size_t NX = (size_t)BATCH * SEQ * DIM;  // 8,388,608
  const size_t NW = (size_t)DIM * DIM;          // 1,048,576

  unsigned short* ws  = (unsigned short*)d_ws;
  unsigned short* Xbf = ws;                      // 16 MB
  unsigned short* Wbf = ws + NX;                 //  6 MB (3 weights)
  unsigned short* Qp  = Wbf + 3 * NW;            // 16 MB
  unsigned short* Kp  = Qp + NX;                 // 16 MB
  unsigned short* Vtp = Kp + NX;                 // 16 MB ([B, D, S])

  convert_bf16<<<2048, 256, 0, stream>>>(X, Xbf, NX);
  convert_bf16<<<512, 256, 0, stream>>>(wq, Wbf, NW);
  convert_bf16<<<512, 256, 0, stream>>>(wk, Wbf + NW, NW);
  convert_bf16<<<512, 256, 0, stream>>>(wv, Wbf + 2 * NW, NW);

  dim3 g1(BATCH * SEQ / 128, DIM / 128, 3);
  qkv_gemm<<<g1, 256, 0, stream>>>(Xbf, Wbf, Qp, Kp, Vtp);

  flash_attn<<<BATCH * (SEQ / 16), 256, 0, stream>>>(Qp, Kp, Vtp,
                                                     (float*)d_out);
}
